// Encoder_29824252903582
// MI455X (gfx1250) — matrix operand order repaired and checked by reading, not compiled, not run on hardware
//
#include <hip/hip_runtime.h>
#include <hip/hip_bf16.h>

typedef __attribute__((ext_vector_type(16))) __bf16 v16bf;
typedef __attribute__((ext_vector_type(8)))  __bf16 v8bf;
typedef __attribute__((ext_vector_type(8)))  float  v8f;
typedef __attribute__((ext_vector_type(4)))  int    v4i;

__device__ __forceinline__ __bf16 f2bf(float f) { return (__bf16)f; }

// Pack f32 weights [R, cout] -> bf16 fragment order pk[(kc*cout + co)*32 + K].
// Rows rb >= R (zero-pad up to Rp, multiple of 32) are written as 0.
__global__ void pack_w_bf16(const float* __restrict__ w, __bf16* __restrict__ pk,
                            int R, int Rp, int cout, int coutSh)
{
  int idx = blockIdx.x * 256 + threadIdx.x;          // over Rp*cout
  if (idx >= Rp * cout) return;
  int K    = idx & 31;
  int rest = idx >> 5;                               // kc*cout + co
  int co   = rest & (cout - 1);
  int kc   = rest >> coutSh;
  int rb   = kc * 32 + K;
  pk[idx] = f2bf(rb < R ? w[rb * cout + co] : 0.0f);
}

// Common per-wave tile geometry (M = 32 voxels).
#define CONV_VOX_SETUP()                                                      \
  const int lane = threadIdx.x & 31;                                          \
  const int wv   = threadIdx.x >> 5;                                          \
  const int mrow = lane & 15;                                                 \
  const int half = lane >> 4;                                                 \
  const int h8   = half * 8;                                                  \
  const int tb = blockIdx.x * 32;                                             \
  int  t0 = tb + mrow, t1 = tb + 16 + mrow;                                   \
  int  bb[2], zo[2], yo[2], xo[2];                                            \
  bool tv[2] = { t0 < nvox, t1 < nvox };                                      \
  {                                                                           \
    int ts[2] = { tv[0] ? t0 : 0, tv[1] ? t1 : 0 };                           \
    _Pragma("unroll")                                                         \
    for (int i = 0; i < 2; ++i) {                                             \
      bb[i] = ts[i] >> (3 * doSh);                                            \
      int rem = ts[i] & ((1 << (3 * doSh)) - 1);                              \
      zo[i] = rem >> (2 * doSh);                                              \
      yo[i] = (rem >> doSh) & (Do - 1);                                       \
      xo[i] = rem & (Do - 1);                                                 \
    }                                                                         \
  }

// ---------------------------------------------------------------------------
// FAST: cin >= 32 (multiple of 32), cout multiple of 32. bf16 activations in,
// packed bf16 weights. Wave tile = 32 voxels x 32 couts -> 4 WMMAs per chunk
// from 4 A-b128 + 4 B-b128 loads (A and B fragments each reused twice).
// Tap-loop live state is just {flat base offset, 27-bit validity mask} per
// row: per-tap address delta is wave-uniform SALU; invalid taps load nearby
// in-workspace garbage that is AND-killed before the WMMA (no clamping).
// ---------------------------------------------------------------------------
__global__ void conv3d_wmma_fast(
    const __bf16* __restrict__ in, const __bf16* __restrict__ wpk,
    float* __restrict__ out,
    int nvox, int Do, int doSh, int Di, int cin, int cout, int coutSh, int stride)
{
  CONV_VOX_SETUP();
  const int co0 = blockIdx.y * blockDim.x + wv * 32 + mrow;   // always < cout
  const int co1 = co0 + 16;

  int      pofs[2];
  unsigned vmask[2];
  #pragma unroll
  for (int i = 0; i < 2; ++i) {
    int base = ((bb[i] * Di + zo[i] * stride) * Di + yo[i] * stride) * Di
               + xo[i] * stride;
    pofs[i] = base * cin + h8;
    unsigned zm = 0, ym = 0, xm = 0;
    #pragma unroll
    for (int k = 0; k < 3; ++k) {
      zm |= (unsigned)((unsigned)(zo[i] * stride + k - 1) < (unsigned)Di) << k;
      ym |= (unsigned)((unsigned)(yo[i] * stride + k - 1) < (unsigned)Di) << k;
      xm |= (unsigned)((unsigned)(xo[i] * stride + k - 1) < (unsigned)Di) << k;
    }
    unsigned yx = 0;
    #pragma unroll
    for (int ky = 0; ky < 3; ++ky)
      #pragma unroll
      for (int kx = 0; kx < 3; ++kx)
        yx |= (((ym >> ky) & (xm >> kx)) & 1u) << (ky * 3 + kx);
    unsigned vm = 0;
    #pragma unroll
    for (int kz = 0; kz < 3; ++kz)
      vm |= ((zm >> kz) & 1u) ? (yx << (9 * kz)) : 0u;
    vmask[i] = tv[i] ? vm : 0u;
  }

  v8f acc00, acc01, acc10, acc11;
  #pragma unroll
  for (int i = 0; i < 8; ++i) {
    acc00[i] = 0.0f; acc01[i] = 0.0f; acc10[i] = 0.0f; acc11[i] = 0.0f;
  }

  for (int tap = 0; tap < 27; ++tap) {
    const int kz = tap / 9;
    const int kr = tap - kz * 9;
    const int ky = kr / 3;
    const int kx = kr - ky * 3;
    const int doff = (((kz - 1) * Di + (ky - 1)) * Di + (kx - 1)) * cin; // uniform
    const int km0 = -(int)((vmask[0] >> tap) & 1u);
    const int km1 = -(int)((vmask[1] >> tap) & 1u);

    for (int cb = 0; cb < cin; cb += 32) {
      v16bf a0, a1;
      {
        const __bf16* p = in + pofs[0] + doff + cb;
        v4i r0 = *(const v4i*)(p);            // K = h8 .. h8+7
        v4i r1 = *(const v4i*)(p + 16);       // K = h8+16 .. h8+23
        r0 &= km0;
        r1 &= km0;
        a0 = __builtin_shufflevector(__builtin_bit_cast(v8bf, r0),
                                     __builtin_bit_cast(v8bf, r1),
                                     0, 1, 2, 3, 4, 5, 6, 7,
                                     8, 9, 10, 11, 12, 13, 14, 15);
      }
      {
        const __bf16* p = in + pofs[1] + doff + cb;
        v4i r0 = *(const v4i*)(p);
        v4i r1 = *(const v4i*)(p + 16);
        r0 &= km1;
        r1 &= km1;
        a1 = __builtin_shufflevector(__builtin_bit_cast(v8bf, r0),
                                     __builtin_bit_cast(v8bf, r1),
                                     0, 1, 2, 3, 4, 5, 6, 7,
                                     8, 9, 10, 11, 12, 13, 14, 15);
      }

      const int kc = (tap * cin + cb) >> 5;
      const __bf16* wb = wpk + ((kc << coutSh) << 5) + half * 8;
      const v8bf* bp0 = (const v8bf*)(wb + (co0 << 5));
      const v8bf* bp1 = (const v8bf*)(wb + (co1 << 5));
      v16bf bm0 = __builtin_shufflevector(bp0[0], bp0[2],
          0, 1, 2, 3, 4, 5, 6, 7, 8, 9, 10, 11, 12, 13, 14, 15);
      v16bf bm1 = __builtin_shufflevector(bp1[0], bp1[2],
          0, 1, 2, 3, 4, 5, 6, 7, 8, 9, 10, 11, 12, 13, 14, 15);

      acc00 = __builtin_amdgcn_wmma_f32_16x16x32_bf16(
          false, a0, false, bm0, (short)0, acc00, false, false);
      acc10 = __builtin_amdgcn_wmma_f32_16x16x32_bf16(
          false, a1, false, bm0, (short)0, acc10, false, false);
      acc01 = __builtin_amdgcn_wmma_f32_16x16x32_bf16(
          false, a0, false, bm1, (short)0, acc01, false, false);
      acc11 = __builtin_amdgcn_wmma_f32_16x16x32_bf16(
          false, a1, false, bm1, (short)0, acc11, false, false);
    }
  }

  #pragma unroll
  for (int v = 0; v < 8; ++v) {
    int row = v + h8;
    int ta  = tb + row;
    int tc  = tb + 16 + row;
    if (ta < nvox) {
      out[ta * cout + co0] = acc00[v];
      out[ta * cout + co1] = acc01[v];
    }
    if (tc < nvox) {
      out[tc * cout + co0] = acc10[v];
      out[tc * cout + co1] = acc11[v];
    }
  }
}

// ---------------------------------------------------------------------------
// C16: cin == 16, bf16 activations, 16-cout tile. Linear K-chunks: chunk c =
// taps {2c,2c+1}. Element block [0..7] <- tap0 @ ci=h8+e, [8..15] <- tap1.
// ---------------------------------------------------------------------------
__global__ void conv3d_wmma_c16(
    const __bf16* __restrict__ in, const __bf16* __restrict__ wpk,
    float* __restrict__ out,
    int nvox, int Do, int doSh, int Di, int cout, int coutSh, int stride)
{
  CONV_VOX_SETUP();
  const int co   = blockIdx.y * 64 + wv * 16 + mrow;
  const bool cvalid = (co < cout);
  const int coc  = cvalid ? co : 0;

  v8f acc0, acc1;
  #pragma unroll
  for (int i = 0; i < 8; ++i) { acc0[i] = 0.0f; acc1[i] = 0.0f; }

  for (int c = 0; c < 14; ++c) {             // Rp = 448 = 14*32, R = 432
    int vb[2][2], km[2][2];
    #pragma unroll
    for (int j = 0; j < 2; ++j) {
      int tap = 2 * c + j;
      int kz = tap / 9;
      int kr = tap - kz * 9;
      int ky = kr / 3;
      int kx = kr - ky * 3;
      bool tok = tap < 27;
      #pragma unroll
      for (int i = 0; i < 2; ++i) {
        int zi = zo[i] * stride + kz - 1;
        int yi = yo[i] * stride + ky - 1;
        int xi = xo[i] * stride + kx - 1;
        bool ok = tok && tv[i] && (unsigned)zi < (unsigned)Di &&
                  (unsigned)yi < (unsigned)Di && (unsigned)xi < (unsigned)Di;
        int vi = ok ? (((bb[i] * Di + zi) * Di + yi) * Di + xi) : 0;
        vb[i][j] = vi * 16;
        km[i][j] = ok ? -1 : 0;
      }
    }

    v16bf a0, a1;
    #pragma unroll
    for (int i = 0; i < 2; ++i) {
      v4i r0 = *(const v4i*)(in + vb[i][0] + h8);
      v4i r1 = *(const v4i*)(in + vb[i][1] + h8);
      r0 &= km[i][0];
      r1 &= km[i][1];
      v16bf& A = i ? a1 : a0;
      A = __builtin_shufflevector(__builtin_bit_cast(v8bf, r0),
                                  __builtin_bit_cast(v8bf, r1),
                                  0, 1, 2, 3, 4, 5, 6, 7,
                                  8, 9, 10, 11, 12, 13, 14, 15);
    }

    const v8bf* bp = (const v8bf*)(wpk + (((c << coutSh) + coc) << 5) + half * 8);
    v16bf bm = __builtin_shufflevector(bp[0], bp[2],
        0, 1, 2, 3, 4, 5, 6, 7, 8, 9, 10, 11, 12, 13, 14, 15);

    acc0 = __builtin_amdgcn_wmma_f32_16x16x32_bf16(
        false, a0, false, bm, (short)0, acc0, false, false);
    acc1 = __builtin_amdgcn_wmma_f32_16x16x32_bf16(
        false, a1, false, bm, (short)0, acc1, false, false);
  }

  if (cvalid) {
    #pragma unroll
    for (int v = 0; v < 8; ++v) {
      int row = v + h8;
      int ta  = tb + row;
      int tc  = tb + 16 + row;
      if (ta < nvox) out[ta * cout + co] = acc0[v];
      if (tc < nvox) out[tc * cout + co] = acc1[v];
    }
  }
}

// ---------------------------------------------------------------------------
// C8: cin == 8 (level 0 conv1 only). f32 input x + occupancy mask folded into
// the gather. Linear K-chunks: chunk c = taps {4c..4c+3}; per lane, element
// block [0..7] <- tap (4c+half) @ ci=e, block [8..15] <- tap (4c+2+half).
// ---------------------------------------------------------------------------
__global__ void conv3d_wmma_c8(
    const float* __restrict__ in, const float* __restrict__ in_mask,
    const __bf16* __restrict__ wpk, float* __restrict__ out,
    int nvox, int Do, int doSh, int Di, int cout, int coutSh, int stride)
{
  CONV_VOX_SETUP();
  const int co   = blockIdx.y * 64 + wv * 16 + mrow;
  const bool cvalid = (co < cout);
  const int coc  = cvalid ? co : 0;

  v8f acc0, acc1;
  #pragma unroll
  for (int i = 0; i < 8; ++i) { acc0[i] = 0.0f; acc1[i] = 0.0f; }

  for (int c = 0; c < 7; ++c) {              // Rp = 224 = 7*32, R = 216
    int   vb[2][4];
    float mv[2][4];
    #pragma unroll
    for (int j = 0; j < 4; ++j) {
      int tap = 4 * c + j;
      int kz = tap / 9;
      int kr = tap - kz * 9;
      int ky = kr / 3;
      int kx = kr - ky * 3;
      bool tok = tap < 27;
      #pragma unroll
      for (int i = 0; i < 2; ++i) {
        int zi = zo[i] * stride + kz - 1;
        int yi = yo[i] * stride + ky - 1;
        int xi = xo[i] * stride + kx - 1;
        bool ok = tok && tv[i] && (unsigned)zi < (unsigned)Di &&
                  (unsigned)yi < (unsigned)Di && (unsigned)xi < (unsigned)Di;
        int vi = ok ? (((bb[i] * Di + zi) * Di + yi) * Di + xi) : 0;
        vb[i][j] = vi * 8;
        mv[i][j] = ok ? in_mask[vi] : 0.0f;   // h = x*mask folded in; kills OOB
      }
    }

    v16bf a0, a1;
    #pragma unroll
    for (int i = 0; i < 2; ++i) {
      int   vb0 = half ? vb[i][1] : vb[i][0];
      int   vb1 = half ? vb[i][3] : vb[i][2];
      float s0  = half ? mv[i][1] : mv[i][0];
      float s1  = half ? mv[i][3] : mv[i][2];
      const float* p0 = in + vb0;
      const float* p1 = in + vb1;
      float4 q0 = *(const float4*)(p0);
      float4 q1 = *(const float4*)(p0 + 4);
      float4 q2 = *(const float4*)(p1);
      float4 q3 = *(const float4*)(p1 + 4);
      v16bf& A = i ? a1 : a0;
      A[0]  = f2bf(q0.x * s0); A[1]  = f2bf(q0.y * s0);
      A[2]  = f2bf(q0.z * s0); A[3]  = f2bf(q0.w * s0);
      A[4]  = f2bf(q1.x * s0); A[5]  = f2bf(q1.y * s0);
      A[6]  = f2bf(q1.z * s0); A[7]  = f2bf(q1.w * s0);
      A[8]  = f2bf(q2.x * s1); A[9]  = f2bf(q2.y * s1);
      A[10] = f2bf(q2.z * s1); A[11] = f2bf(q2.w * s1);
      A[12] = f2bf(q3.x * s1); A[13] = f2bf(q3.y * s1);
      A[14] = f2bf(q3.z * s1); A[15] = f2bf(q3.w * s1);
    }

    const v8bf* bp = (const v8bf*)(wpk + (((c << coutSh) + coc) << 5) + half * 8);
    v16bf bm = __builtin_shufflevector(bp[0], bp[2],
        0, 1, 2, 3, 4, 5, 6, 7, 8, 9, 10, 11, 12, 13, 14, 15);

    acc0 = __builtin_amdgcn_wmma_f32_16x16x32_bf16(
        false, a0, false, bm, (short)0, acc0, false, false);
    acc1 = __builtin_amdgcn_wmma_f32_16x16x32_bf16(
        false, a1, false, bm, (short)0, acc1, false, false);
  }

  if (cvalid) {
    #pragma unroll
    for (int v = 0; v < 8; ++v) {
      int row = v + h8;
      int ta  = tb + row;
      int tc  = tb + 16 + row;
      if (ta < nvox) out[ta * cout + co] = acc0[v];
      if (tc < nvox) out[tc * cout + co] = acc1[v];
    }
  }
}

// 2x2x2 occupancy max-pool (mask pyramid).
__global__ void pool_mask_k(const float* __restrict__ mi, float* __restrict__ mo,
                            int nvox, int Do, int doSh)
{
  int t = blockIdx.x * blockDim.x + threadIdx.x;
  if (t >= nvox) return;
  int b   = t >> (3 * doSh);
  int rem = t & ((1 << (3 * doSh)) - 1);
  int zo = rem >> (2 * doSh), yo = (rem >> doSh) & (Do - 1), xo = rem & (Do - 1);
  int Di = Do * 2;
  float mx = 0.0f;
  #pragma unroll
  for (int dz = 0; dz < 2; ++dz)
    #pragma unroll
    for (int dy = 0; dy < 2; ++dy)
      #pragma unroll
      for (int dx = 0; dx < 2; ++dx) {
        int vi = (((b * Di + (2 * zo + dz)) * Di + (2 * yo + dy)) * Di + (2 * xo + dx));
        mx = fmaxf(mx, mi[vi]);
      }
  mo[t] = mx;
}

__global__ void mask_count_k(const float* __restrict__ m, float* __restrict__ cnt, int nvox)
{
  __shared__ float s[256];
  float a = 0.0f;
  for (int i = threadIdx.x; i < nvox; i += 256) a += m[i];
  s[threadIdx.x] = a;
  __syncthreads();
  for (int st = 128; st > 0; st >>= 1) {
    if (threadIdx.x < st) s[threadIdx.x] += s[threadIdx.x + st];
    __syncthreads();
  }
  if (threadIdx.x == 0) cnt[0] = fmaxf(s[0], 1.0f);
}

__global__ void bn_stats_k(const float* __restrict__ h, const float* __restrict__ m,
                           float* __restrict__ sum, float* __restrict__ sumsq,
                           int nvox, int cout)
{
  __shared__ float s1[256], s2[256];
  int c = blockIdx.x;
  float a1 = 0.0f, a2 = 0.0f;
  for (int v = threadIdx.x; v < nvox; v += 256) {
    float x  = h[v * cout + c];
    float mm = m[v];
    a1 += x * mm;
    a2 += x * x * mm;
  }
  s1[threadIdx.x] = a1; s2[threadIdx.x] = a2;
  __syncthreads();
  for (int st = 128; st > 0; st >>= 1) {
    if (threadIdx.x < st) { s1[threadIdx.x] += s1[threadIdx.x + st];
                            s2[threadIdx.x] += s2[threadIdx.x + st]; }
    __syncthreads();
  }
  if (threadIdx.x == 0) { sum[c] = s1[0]; sumsq[c] = s2[0]; }
}

// Masked BN + ELU, f32 conv output -> bf16 activations for the next conv.
__global__ void bn_apply_k(const float* __restrict__ h, __bf16* __restrict__ ho,
                           const float* __restrict__ m,
                           const float* __restrict__ sum, const float* __restrict__ sumsq,
                           const float* __restrict__ cnt,
                           const float* __restrict__ gamma, const float* __restrict__ beta,
                           int nvox, int cout, int coutSh)
{
  int idx   = blockIdx.x * blockDim.x + threadIdx.x;
  int total = nvox << coutSh;
  if (idx >= total) return;
  int c = idx & (cout - 1);
  int v = idx >> coutSh;
  float n   = cnt[0];
  float mu  = sum[c] / n;
  float var = sumsq[c] / n - mu * mu;
  float x = h[idx];
  float y = ((x - mu) * rsqrtf(var + 1e-5f) * gamma[c] + beta[c]) * m[v];
  ho[idx] = f2bf(y > 0.0f ? y : (expf(y) - 1.0f));
}

__global__ void head_k(const __bf16* __restrict__ h, const float* __restrict__ m,
                       const float* __restrict__ W, const float* __restrict__ bias,
                       float* __restrict__ out)
{
  int idx = blockIdx.x * blockDim.x + threadIdx.x;  // 0..2047
  if (idx >= 2048) return;
  int b = idx >> 10, j = idx & 1023;
  float mm = m[b];
  float p  = mm / fmaxf(mm, 1.0f);
  float acc = bias[j];
  for (int c = 0; c < 1024; ++c)
    acc += p * (float)h[b * 1024 + c] * W[c * 1024 + j];
  out[idx] = acc;
}

extern "C" void kernel_launch(void* const* d_in, const int* in_sizes, int n_in,
                              void* d_out, int out_size, void* d_ws, size_t ws_size,
                              hipStream_t stream)
{
  (void)in_sizes; (void)n_in; (void)out_size; (void)ws_size;
  const float* x    = (const float*)d_in[0];
  const float* mask = (const float*)d_in[1];
  const float *w1[7], *w2[7], *g1[7], *g2[7], *bb1[7], *bb2[7];
  for (int l = 0; l < 7; ++l) {
    w1[l]  = (const float*)d_in[2  + 2 * l];
    w2[l]  = (const float*)d_in[3  + 2 * l];
    g1[l]  = (const float*)d_in[16 + 2 * l];
    g2[l]  = (const float*)d_in[17 + 2 * l];
    bb1[l] = (const float*)d_in[30 + 2 * l];
    bb2[l] = (const float*)d_in[31 + 2 * l];
  }
  const float* Wm   = (const float*)d_in[44];
  const float* bias = (const float*)d_in[45];

  float*  ws   = (float*)d_ws;
  float*  bufC = ws;                           // f32 conv output (max 8.39M)
  __bf16* hb0  = (__bf16*)(ws + 8388608);      // bf16 activations (ping)
  __bf16* hb1  = (__bf16*)(ws + 12582912);     // bf16 activations (pong)
  static const int msz[8] = {0, 524288, 65536, 8192, 1024, 128, 16, 2};
  float* mptr[8]; mptr[0] = nullptr;
  size_t o = 16777216;
  for (int l = 1; l < 8; ++l) { mptr[l] = ws + o; o += msz[l]; }
  float* stat_sum = ws + o; o += 1024;
  float* stat_sq  = ws + o; o += 1024;
  float* stat_cnt = ws + o; o += 1024;         // padded for wpk alignment
  __bf16* wpk = (__bf16*)(ws + o);             // max 27648*1024 bf16 = 56.6MB

  static const int DiA[7]     = {128, 64, 32, 16, 8, 4, 2};
  static const int DoA[7]     = {64, 32, 16, 8, 4, 2, 1};
  static const int doShA[7]   = {6, 5, 4, 3, 2, 1, 0};
  static const int cinA[7]    = {8, 16, 32, 64, 128, 256, 512};
  static const int coutA[7]   = {16, 32, 64, 128, 256, 512, 1024};
  static const int coutShA[7] = {4, 5, 6, 7, 8, 9, 10};

  // Mask pyramid m1..m7.
  const float* mprev = mask;
  for (int l = 0; l < 7; ++l) {
    int nv = 2 * DoA[l] * DoA[l] * DoA[l];
    pool_mask_k<<<(nv + 255) / 256, 256, 0, stream>>>(mprev, mptr[l + 1], nv, DoA[l], doShA[l]);
    mprev = mptr[l + 1];
  }

  for (int l = 0; l < 7; ++l) {
    int Di = DiA[l], Do = DoA[l], cin = cinA[l], cout = coutA[l], cSh = coutShA[l];
    int nv  = 2 * Do * Do * Do;
    int tot = nv * cout;
    int gx  = (nv + 31) / 32;
    // fast kernel: each wave covers 32 couts -> threads == couts per block
    int thrF = cout >= 128 ? 128 : cout;
    dim3 gridF(gx, cout / thrF);
    // c8/c16 kernels: each wave covers 16 couts
    int thrS = cout >= 64 ? 128 : (cout == 32 ? 64 : 32);
    dim3 gridS(gx, (cout + 63) / 64);

    // ---- conv1 (stride 2) -> bufC ----
    if (l == 0) {
      pack_w_bf16<<<(224 * 16 + 255) / 256, 256, 0, stream>>>(w1[0], wpk, 216, 224, 16, 4);
      conv3d_wmma_c8<<<gridS, thrS, 0, stream>>>(
          x, mask, wpk, bufC, nv, Do, doShA[l], Di, cout, cSh, 2);
    } else if (cin == 16) {
      pack_w_bf16<<<(448 * cout + 255) / 256, 256, 0, stream>>>(w1[l], wpk, 432, 448, cout, cSh);
      conv3d_wmma_c16<<<gridS, thrS, 0, stream>>>(
          hb1, wpk, bufC, nv, Do, doShA[l], Di, cout, cSh, 2);
    } else {
      int R = 27 * cin;
      pack_w_bf16<<<(R * cout + 255) / 256, 256, 0, stream>>>(w1[l], wpk, R, R, cout, cSh);
      conv3d_wmma_fast<<<gridF, thrF, 0, stream>>>(
          hb1, wpk, bufC, nv, Do, doShA[l], Di, cin, cout, cSh, 2);
    }
    mask_count_k<<<1, 256, 0, stream>>>(mptr[l + 1], stat_cnt, nv);
    bn_stats_k<<<cout, 256, 0, stream>>>(bufC, mptr[l + 1], stat_sum, stat_sq, nv, cout);
    bn_apply_k<<<(tot + 255) / 256, 256, 0, stream>>>(
        bufC, hb0, mptr[l + 1], stat_sum, stat_sq, stat_cnt, g1[l], bb1[l], nv, cout, cSh);

    // ---- conv2 (stride 1, cin == cout) -> bufC ----
    if (cout == 16) {
      pack_w_bf16<<<(448 * 16 + 255) / 256, 256, 0, stream>>>(w2[l], wpk, 432, 448, 16, 4);
      conv3d_wmma_c16<<<gridS, thrS, 0, stream>>>(
          hb0, wpk, bufC, nv, Do, doShA[l], Do, cout, cSh, 1);
    } else {
      int R = 27 * cout;
      pack_w_bf16<<<(R * cout + 255) / 256, 256, 0, stream>>>(w2[l], wpk, R, R, cout, cSh);
      conv3d_wmma_fast<<<gridF, thrF, 0, stream>>>(
          hb0, wpk, bufC, nv, Do, doShA[l], Do, cout, cout, cSh, 1);
    }
    bn_stats_k<<<cout, 256, 0, stream>>>(bufC, mptr[l + 1], stat_sum, stat_sq, nv, cout);
    bn_apply_k<<<(tot + 255) / 256, 256, 0, stream>>>(
        bufC, hb1, mptr[l + 1], stat_sum, stat_sq, stat_cnt, g2[l], bb2[l], nv, cout, cSh);
  }

  head_k<<<8, 256, 0, stream>>>(hb1, mptr[7], Wm, bias, (float*)d_out);
}
